// CustomizedNeighborhoodAttention1D_MH_72679436583281
// MI455X (gfx1250) — matrix operand order repaired and checked by reading, not compiled, not run on hardware
//
#include <hip/hip_runtime.h>
#include <hip/hip_bf16.h>

typedef _Float16 half_t;
typedef __attribute__((ext_vector_type(16))) _Float16 v16h;
typedef __attribute__((ext_vector_type(8)))  _Float16 v8h;
typedef __attribute__((ext_vector_type(8)))  float    v8f;

union VU  { v16h v; v8h h8[2]; uint4 u[2]; };
union PK8 { v8h h; uint4 u; };

static __device__ inline uint4 u4zero() { uint4 z; z.x = z.y = z.z = z.w = 0u; return z; }

// ---------------- conversion kernels ----------------

__global__ void cvt_f16_kernel(const float* __restrict__ s, half_t* __restrict__ d, int n) {
    int i = blockIdx.x * blockDim.x + threadIdx.x;
    int stride = gridDim.x * blockDim.x;
    for (; i < n; i += stride) d[i] = (half_t)s[i];
}

// d[n*K + k] = (half) s[k*N + n]   (store W transposed, N-major rows)
__global__ void transpose_f16_kernel(const float* __restrict__ s, half_t* __restrict__ d, int K, int N) {
    int i = blockIdx.x * blockDim.x + threadIdx.x;
    int stride = gridDim.x * blockDim.x;
    int total = K * N;
    for (; i < total; i += stride) {
        int nn = i / K, kk = i - nn * K;
        d[i] = (half_t)s[kk * N + nn];
    }
}

// ---------------- QKV GEMM: (16384x256) x (256x768) + bias ----------------
// One 16(M) x 64(N) tile per wave: 4 accumulators share one A fragment.
// Q,K -> (bh, l, 32) f16 row-major; V -> (bh, 32, L) f16 transposed.

__global__ void qkv_gemm_kernel(const half_t* __restrict__ Xh, const half_t* __restrict__ Wt,
                                const float* __restrict__ bias_qkv,
                                half_t* __restrict__ Qh, half_t* __restrict__ Kh,
                                half_t* __restrict__ Vt) {
    const int lane = threadIdx.x & 31;
    const int wave = threadIdx.x >> 5;
    const int m0 = (blockIdx.x * 8 + wave) * 16;     // row tile (over B*L = 16384)
    const int n0 = blockIdx.y * 64;                  // col super-tile (over 768)
    const int n   = lane & 15;
    const int sel = lane >> 4;

    const half_t* arow = Xh + (size_t)(m0 + n) * 256;            // A: lane = row m
    const half_t* brow0 = Wt + (size_t)(n0 + n) * 256;           // B tile j: lane = col n0+16j+n
    const half_t* brow1 = Wt + (size_t)(n0 + 16 + n) * 256;
    const half_t* brow2 = Wt + (size_t)(n0 + 32 + n) * 256;
    const half_t* brow3 = Wt + (size_t)(n0 + 48 + n) * 256;

    __builtin_prefetch(arow, 0, 1);                  // global_prefetch_b8
    __builtin_prefetch(brow0, 0, 1);
    __builtin_prefetch(brow2, 0, 1);

    v8f acc0 = {}, acc1 = {}, acc2 = {}, acc3 = {};
    for (int k0 = 0; k0 < 256; k0 += 32) {
        VU a, b0, b1, b2, b3;
        a.u[0]  = *(const uint4*)(arow + k0 + 16 * sel);
        a.u[1]  = *(const uint4*)(arow + k0 + 16 * sel + 8);
        b0.u[0] = *(const uint4*)(brow0 + k0 + 16 * sel);
        b0.u[1] = *(const uint4*)(brow0 + k0 + 16 * sel + 8);
        b1.u[0] = *(const uint4*)(brow1 + k0 + 16 * sel);
        b1.u[1] = *(const uint4*)(brow1 + k0 + 16 * sel + 8);
        b2.u[0] = *(const uint4*)(brow2 + k0 + 16 * sel);
        b2.u[1] = *(const uint4*)(brow2 + k0 + 16 * sel + 8);
        b3.u[0] = *(const uint4*)(brow3 + k0 + 16 * sel);
        b3.u[1] = *(const uint4*)(brow3 + k0 + 16 * sel + 8);
        acc0 = __builtin_amdgcn_wmma_f32_16x16x32_f16(false, a.v, false, b0.v, (short)0, acc0, false, false);
        acc1 = __builtin_amdgcn_wmma_f32_16x16x32_f16(false, a.v, false, b1.v, (short)0, acc1, false, false);
        acc2 = __builtin_amdgcn_wmma_f32_16x16x32_f16(false, a.v, false, b2.v, (short)0, acc2, false, false);
        acc3 = __builtin_amdgcn_wmma_f32_16x16x32_f16(false, a.v, false, b3.v, (short)0, acc3, false, false);
    }

    const int b_idx = m0 >> 13;                      // batch (L = 8192)
    const int l_base = (m0 & 8191) + 8 * sel;

    v8f accs[4] = {acc0, acc1, acc2, acc3};
    for (int j = 0; j < 4; ++j) {
        const int c0 = n0 + 16 * j;                  // column tile base, multiple of 16
        const float bias = bias_qkv[c0 + n];
        const int type = c0 >> 8;                    // 0=Q 1=K 2=V (64-supertile never straddles)
        const int h    = (c0 & 255) >> 5;            // head
        const int d    = (c0 & 31) + n;              // 0..31 within head
        const int bh   = b_idx * 8 + h;
        if (type == 2) {
            PK8 pk;
            for (int r = 0; r < 8; ++r) pk.h[r] = (half_t)(accs[j][r] + bias);
            *(uint4*)(Vt + ((size_t)bh * 32 + d) * 8192 + l_base) = pk.u;
        } else {
            half_t* dst = (type == 0 ? Qh : Kh);
            for (int r = 0; r < 8; ++r) {
                int l = l_base + r;
                dst[((size_t)bh * 8192 + l) * 32 + d] = (half_t)(accs[j][r] + bias);
            }
        }
    }
}

// ---------------- neighborhood attention: 16 queries per wave ----------------

__global__ void attn_kernel(const half_t* __restrict__ Qh, const half_t* __restrict__ Kh,
                            const half_t* __restrict__ Vt, half_t* __restrict__ Ah) {
    __shared__ __align__(16) half_t Pl[8][16 * 96];

    const int lane = threadIdx.x & 31;
    const int wave = threadIdx.x >> 5;
    const int w  = blockIdx.x * 8 + wave;   // 0..8191
    const int bh = w >> 9;                  // 16 (b,h) pairs
    const int l0 = (w & 511) << 4;          // query tile start
    const int n   = lane & 15;
    const int sel = lane >> 4;

    // A fragment: 16 query rows of head_dim 32
    VU qa;
    {
        const half_t* qrow = Qh + ((size_t)bh * 8192 + (l0 + n)) * 32;
        qa.u[0] = *(const uint4*)(qrow + 16 * sel);
        qa.u[1] = *(const uint4*)(qrow + 16 * sel + 8);
    }

    // warm V rows for the PV phase (speculative prefetch, clamped)
    {
        int kc = l0 - 32 < 0 ? 0 : l0 - 32;
        __builtin_prefetch(Vt + ((size_t)bh * 32 + n) * 8192 + kc, 0, 1);
        __builtin_prefetch(Vt + ((size_t)bh * 32 + 16 + n) * 8192 + kc, 0, 1);
    }

    // zero P padding cols [80,96)
    {
        int zr = lane >> 1, zc = 80 + 8 * (lane & 1);
        *(uint4*)&Pl[wave][zr * 96 + zc] = u4zero();
    }

    const float scale = 0.17677669529663687f;  // 1/sqrt(32)
    v8f s[5];
    for (int t = 0; t < 5; ++t) {
        VU kb;
        int kpos = l0 - 32 + 16 * t + n;       // absolute key index for this lane's column
        if (kpos >= 0 && kpos < 8192) {
            const half_t* krow = Kh + ((size_t)bh * 8192 + kpos) * 32 + 16 * sel;
            kb.u[0] = *(const uint4*)(krow);
            kb.u[1] = *(const uint4*)(krow + 8);
        } else {
            kb.u[0] = u4zero(); kb.u[1] = u4zero();
        }
        v8f z = {};
        z = __builtin_amdgcn_wmma_f32_16x16x32_f16(false, qa.v, false, kb.v,
                                                   (short)0, z, false, false);
        for (int r = 0; r < 8; ++r) {
            int i = r + 8 * sel;               // query row within tile
            int c = 16 * t + n;                // window column 0..79
            float val = z[r] * scale;
            if (c < i || c > i + 64) val = -1e30f;   // outside 65-wide band
            s[t][r] = val;
        }
    }

    // row max across tiles + 16-lane half-group
    v8f rm;
    for (int r = 0; r < 8; ++r) {
        float m = s[0][r];
        for (int t = 1; t < 5; ++t) m = fmaxf(m, s[t][r]);
        rm[r] = m;
    }
    for (int off = 1; off < 16; off <<= 1)
        for (int r = 0; r < 8; ++r)
            rm[r] = fmaxf(rm[r], __shfl_xor(rm[r], off, 32));

    // exp + row sum
    v8f rs;
    for (int r = 0; r < 8; ++r) rs[r] = 0.f;
    for (int t = 0; t < 5; ++t)
        for (int r = 0; r < 8; ++r) {
            float e = __expf(s[t][r] - rm[r]);
            s[t][r] = e;
            rs[r] += e;
        }
    for (int off = 1; off < 16; off <<= 1)
        for (int r = 0; r < 8; ++r)
            rs[r] += __shfl_xor(rs[r], off, 32);

    // normalized P -> LDS (row-major 16x96, cols 80..95 already zero)
    for (int t = 0; t < 5; ++t)
        for (int r = 0; r < 8; ++r) {
            int i = r + 8 * sel, c = 16 * t + n;
            Pl[wave][i * 96 + c] = (half_t)(s[t][r] / rs[r]);
        }
    __syncthreads();

    // O = P (16x96, zero-padded) x V (96x32): 3 K-chunks x 2 N-tiles
    v8f o0 = {}, o1 = {};
    for (int j = 0; j < 3; ++j) {
        VU pa;   // A fragment of P: lane = query row
        pa.u[0] = *(const uint4*)&Pl[wave][n * 96 + 32 * j + 16 * sel];
        pa.u[1] = *(const uint4*)&Pl[wave][n * 96 + 32 * j + 16 * sel + 8];
        int kbase = l0 - 32 + 32 * j + 16 * sel;   // 8-aligned key group start
        for (int nt = 0; nt < 2; ++nt) {
            VU vb;   // B fragment from transposed V: lane = output dim column
            const half_t* vrow = Vt + ((size_t)bh * 32 + nt * 16 + n) * 8192;
            if (kbase >= 0 && kbase + 8 <= 8192) vb.u[0] = *(const uint4*)(vrow + kbase);
            else                                  vb.u[0] = u4zero();
            if (kbase + 8 >= 0 && kbase + 16 <= 8192) vb.u[1] = *(const uint4*)(vrow + kbase + 8);
            else                                       vb.u[1] = u4zero();
            if (nt == 0)
                o0 = __builtin_amdgcn_wmma_f32_16x16x32_f16(false, pa.v, false, vb.v,
                                                            (short)0, o0, false, false);
            else
                o1 = __builtin_amdgcn_wmma_f32_16x16x32_f16(false, pa.v, false, vb.v,
                                                            (short)0, o1, false, false);
        }
    }

    // store O as f16 into (B*L, 256) for the projection GEMM
    const int b_idx = bh >> 3, h = bh & 7;
    for (int r = 0; r < 8; ++r) {
        int l = l0 + r + 8 * sel;
        size_t base = ((size_t)b_idx * 8192 + l) * 256 + h * 32;
        Ah[base + n]      = (half_t)o0[r];
        Ah[base + 16 + n] = (half_t)o1[r];
    }
}

// ---------------- projection GEMM: (16384x256) x (256x256) + bias -> f32 out ----------------
// 16(M) x 64(N) per wave, same structure as QKV GEMM.

__global__ void proj_gemm_kernel(const half_t* __restrict__ Ah, const half_t* __restrict__ Wt,
                                 const float* __restrict__ bias_p, float* __restrict__ out) {
    const int lane = threadIdx.x & 31;
    const int wave = threadIdx.x >> 5;
    const int m0 = (blockIdx.x * 8 + wave) * 16;
    const int n0 = blockIdx.y * 64;
    const int n   = lane & 15;
    const int sel = lane >> 4;

    const half_t* arow = Ah + (size_t)(m0 + n) * 256;
    const half_t* brow0 = Wt + (size_t)(n0 + n) * 256;
    const half_t* brow1 = Wt + (size_t)(n0 + 16 + n) * 256;
    const half_t* brow2 = Wt + (size_t)(n0 + 32 + n) * 256;
    const half_t* brow3 = Wt + (size_t)(n0 + 48 + n) * 256;

    __builtin_prefetch(arow, 0, 1);
    __builtin_prefetch(brow0, 0, 1);
    __builtin_prefetch(brow2, 0, 1);

    v8f acc0 = {}, acc1 = {}, acc2 = {}, acc3 = {};
    for (int k0 = 0; k0 < 256; k0 += 32) {
        VU a, b0, b1, b2, b3;
        a.u[0]  = *(const uint4*)(arow + k0 + 16 * sel);
        a.u[1]  = *(const uint4*)(arow + k0 + 16 * sel + 8);
        b0.u[0] = *(const uint4*)(brow0 + k0 + 16 * sel);
        b0.u[1] = *(const uint4*)(brow0 + k0 + 16 * sel + 8);
        b1.u[0] = *(const uint4*)(brow1 + k0 + 16 * sel);
        b1.u[1] = *(const uint4*)(brow1 + k0 + 16 * sel + 8);
        b2.u[0] = *(const uint4*)(brow2 + k0 + 16 * sel);
        b2.u[1] = *(const uint4*)(brow2 + k0 + 16 * sel + 8);
        b3.u[0] = *(const uint4*)(brow3 + k0 + 16 * sel);
        b3.u[1] = *(const uint4*)(brow3 + k0 + 16 * sel + 8);
        acc0 = __builtin_amdgcn_wmma_f32_16x16x32_f16(false, a.v, false, b0.v, (short)0, acc0, false, false);
        acc1 = __builtin_amdgcn_wmma_f32_16x16x32_f16(false, a.v, false, b1.v, (short)0, acc1, false, false);
        acc2 = __builtin_amdgcn_wmma_f32_16x16x32_f16(false, a.v, false, b2.v, (short)0, acc2, false, false);
        acc3 = __builtin_amdgcn_wmma_f32_16x16x32_f16(false, a.v, false, b3.v, (short)0, acc3, false, false);
    }

    v8f accs[4] = {acc0, acc1, acc2, acc3};
    for (int j = 0; j < 4; ++j) {
        const int c = n0 + 16 * j + n;
        const float bias = bias_p[c];
        for (int r = 0; r < 8; ++r) {
            int row = m0 + r + 8 * sel;
            out[(size_t)row * 256 + c] = accs[j][r] + bias;
        }
    }
}

// ---------------- host launcher ----------------

extern "C" void kernel_launch(void* const* d_in, const int* in_sizes, int n_in,
                              void* d_out, int out_size, void* d_ws, size_t ws_size,
                              hipStream_t stream) {
    (void)in_sizes; (void)n_in; (void)out_size; (void)ws_size;
    const float* x      = (const float*)d_in[0];   // (2, 8192, 256)
    const float* w_qkv  = (const float*)d_in[1];   // (256, 768)
    const float* b_qkv  = (const float*)d_in[2];   // (768,)
    const float* w_proj = (const float*)d_in[3];   // (256, 256)
    const float* b_proj = (const float*)d_in[4];   // (256,)
    float* out = (float*)d_out;

    char* ws = (char*)d_ws;
    // Ah aliases Xh: Xh is dead after qkv_gemm, Ah is written later (stream-ordered).
    half_t* Xh     = (half_t*)(ws + 0);            // 16384*256 f16  (8,388,608 B)
    half_t* Ah     = (half_t*)(ws + 0);            // alias
    half_t* WqkvT  = (half_t*)(ws + 8388608);      // 768*256  f16
    half_t* WprojT = (half_t*)(ws + 8781824);      // 256*256  f16
    half_t* Qh     = (half_t*)(ws + 8912896);      // 16*8192*32 f16
    half_t* Kh     = (half_t*)(ws + 17301504);     // 16*8192*32 f16
    half_t* Vt     = (half_t*)(ws + 25690112);     // 16*32*8192 f16 (transposed)
    // total ws use: 34,078,720 bytes

    cvt_f16_kernel<<<8192, 256, 0, stream>>>(x, Xh, 16384 * 256);
    transpose_f16_kernel<<<768, 256, 0, stream>>>(w_qkv, WqkvT, 256, 768);
    transpose_f16_kernel<<<256, 256, 0, stream>>>(w_proj, WprojT, 256, 256);

    qkv_gemm_kernel<<<dim3(128, 12), 256, 0, stream>>>(Xh, WqkvT, b_qkv, Qh, Kh, Vt);
    attn_kernel<<<1024, 256, 0, stream>>>(Qh, Kh, Vt, Ah);
    proj_gemm_kernel<<<dim3(128, 4), 256, 0, stream>>>(Ah, WprojT, b_proj, out);
}
